// SimnetClassifier_72387378807132
// MI455X (gfx1250) — matrix operand order repaired and checked by reading, not compiled, not run on hardware
//
#include <hip/hip_runtime.h>
#include <math.h>

// ---------------------------------------------------------------------------
// Simnet classifier on MI455X (gfx1250), wave32 + WMMA f16 (f32 accum).
//   QA[2048][256] = queries @ W1[:256]          (f16, workspace)
//   SB[ 160][256] = supports @ W1[256:] + b1    (f16, workspace)
//   per (q,c): h1[s] = relu(QA[q] + SB[c*16+s]) ; 16 rows = one WMMA M-tile
//   layers 2..4 via v_wmma_f32_16x16x32_f16, tanh + mean + sum reduction.
// ---------------------------------------------------------------------------

typedef _Float16 f16x8  __attribute__((ext_vector_type(8)));
typedef _Float16 f16x16 __attribute__((ext_vector_type(16)));
typedef float    f32x8  __attribute__((ext_vector_type(8)));

static __device__ __forceinline__ f16x16 cat16(f16x8 lo, f16x8 hi) {
    return __builtin_shufflevector(lo, hi, 0,1,2,3,4,5,6,7,8,9,10,11,12,13,14,15);
}

static __device__ __forceinline__ f16x16 ldB(const _Float16* p) {
    return cat16(*(const f16x8*)p, *(const f16x8*)(p + 16));
}

static __device__ __forceinline__ f32x8 wmma_f16(f16x16 a, f16x16 b, f32x8 c) {
    // D = A(16x32 f16) * B(32x16 f16) + C(16x16 f32)
    return __builtin_amdgcn_wmma_f32_16x16x32_f16(false, a, false, b, (short)0, c,
                                                  false, false);
}

// ---------------------------------------------------------------------------
// Generic [M,256] x [256,256] GEMM, fp32 in, f16 out, optional bias.
// Block = 256 threads (8 waves). W transposed into LDS as f16, padded row
// stride 264 halves (conflict-free B reads). One 16-row tile per wave.
// LDS: WT 256*264*2 = 135168 B, bias 1024 B -> 136192 B dynamic.
// ---------------------------------------------------------------------------
#define P_WT_STRIDE 264
#define P_BIAS_OFF  135168
#define P_SMEM_SIZE 136192

__global__ __launch_bounds__(256, 1)
void gemm256_f16out(const float* __restrict__ A,
                    const float* __restrict__ W,
                    const float* __restrict__ bias,
                    _Float16* __restrict__ out, int M) {
    extern __shared__ __align__(16) char smem[];
    _Float16* WT = (_Float16*)smem;                 // [256][264]
    float*    bl = (float*)(smem + P_BIAS_OFF);     // [256]

    const int tid = threadIdx.x;
    for (int idx = tid; idx < 256 * 256; idx += 256) {
        int n = idx >> 8, k = idx & 255;
        WT[n * P_WT_STRIDE + k] = (_Float16)W[k * 256 + n];
    }
    bl[tid] = bias ? bias[tid] : 0.0f;
    __syncthreads();

    const int lane = tid & 31;
    const int wave = tid >> 5;
    const int half = lane >> 4;   // which K-half this lane carries
    const int lm   = lane & 15;   // row (A) / column (B,D) index
    const int nwaves = gridDim.x * 8;

    const f32x8 vzero = {0.f,0.f,0.f,0.f,0.f,0.f,0.f,0.f};

    for (int rt = blockIdx.x * 8 + wave; rt < (M >> 4); rt += nwaves) {
        const float* ar = A + (size_t)(rt * 16 + lm) * 256;
        for (int nb = 0; nb < 2; ++nb) {            // two blocks of 8 N-tiles
            f32x8 acc[8];
#pragma unroll
            for (int j = 0; j < 8; ++j) acc[j] = vzero;
#pragma unroll
            for (int kt = 0; kt < 8; ++kt) {
                const int k0 = kt * 32 + half * 8;
                float4 f0 = *(const float4*)(ar + k0);
                float4 f1 = *(const float4*)(ar + k0 + 4);
                float4 f2 = *(const float4*)(ar + k0 + 16);
                float4 f3 = *(const float4*)(ar + k0 + 20);
                f16x8 lo = {(_Float16)f0.x,(_Float16)f0.y,(_Float16)f0.z,(_Float16)f0.w,
                            (_Float16)f1.x,(_Float16)f1.y,(_Float16)f1.z,(_Float16)f1.w};
                f16x8 hi = {(_Float16)f2.x,(_Float16)f2.y,(_Float16)f2.z,(_Float16)f2.w,
                            (_Float16)f3.x,(_Float16)f3.y,(_Float16)f3.z,(_Float16)f3.w};
                f16x16 a = cat16(lo, hi);
                // Preload all 8 B operands (independent regs -> clause + overlap)
                f16x16 bmat[8];
#pragma unroll
                for (int j = 0; j < 8; ++j) {
                    const int n = (nb * 8 + j) * 16 + lm;
                    bmat[j] = ldB(WT + n * P_WT_STRIDE + kt * 32 + half * 8);
                }
#pragma unroll
                for (int j = 0; j < 8; ++j)
                    acc[j] = wmma_f16(a, bmat[j], acc[j]);
            }
#pragma unroll
            for (int j = 0; j < 8; ++j) {
                const int n = (nb * 8 + j) * 16 + lm;
                const float bb = bl[n];
#pragma unroll
                for (int v = 0; v < 8; ++v) {
                    const int m = rt * 16 + v + half * 8;
                    out[(size_t)m * 256 + n] = (_Float16)(acc[j][v] + bb);
                }
            }
        }
    }
}

// ---------------------------------------------------------------------------
// Main kernel: one wave per (q,c) tile (16 support rows = M dimension).
// LDS layout (dynamic, 141376 B):
//   W2T [128][264] f16  @      0   (67584 B)
//   W3T [ 64][136] f16  @  67584   (17408 B)
//   W4T [ 16][ 72] f16  @  84992   ( 2304 B)  rows n>=10 zero-padded
//   b2/b3/b4 f32        @  87296 / 87808 / 88064
//   per-wave scratch    @  88128 : h2 [16][136] f16 + h3 [16][72] f16
// ---------------------------------------------------------------------------
#define M_W2T_OFF   0
#define M_W2T_STR   264
#define M_W3T_OFF   67584
#define M_W3T_STR   136
#define M_W4T_OFF   84992
#define M_W4T_STR   72
#define M_B2_OFF    87296
#define M_B3_OFF    87808
#define M_B4_OFF    88064
#define M_SCR_OFF   88128
#define M_H2_STR    136
#define M_H3_STR    72
#define M_SCR_HALVES (16 * M_H2_STR + 16 * M_H3_STR)   // 3328 halves = 6656 B
#define M_SMEM_SIZE (M_SCR_OFF + 8 * M_SCR_HALVES * 2) // 141376 B
#define NTILES      20480                              // 2048 * 10

__global__ __launch_bounds__(256, 1)
void simnet_main(const _Float16* __restrict__ QA,
                 const _Float16* __restrict__ SB,
                 const float* __restrict__ W2, const float* __restrict__ b2,
                 const float* __restrict__ W3, const float* __restrict__ b3,
                 const float* __restrict__ W4, const float* __restrict__ b4,
                 float* __restrict__ out) {
    extern __shared__ __align__(16) char smem[];
    _Float16* W2T = (_Float16*)(smem + M_W2T_OFF);
    _Float16* W3T = (_Float16*)(smem + M_W3T_OFF);
    _Float16* W4T = (_Float16*)(smem + M_W4T_OFF);
    float* b2l = (float*)(smem + M_B2_OFF);
    float* b3l = (float*)(smem + M_B3_OFF);
    float* b4l = (float*)(smem + M_B4_OFF);

    const int tid = threadIdx.x;
    // Stage transposed f16 weights + biases once per block.
    for (int idx = tid; idx < 128 * 256; idx += 256) {        // W2 [256,128]
        int n = idx >> 8, k = idx & 255;
        W2T[n * M_W2T_STR + k] = (_Float16)W2[k * 128 + n];
    }
    for (int idx = tid; idx < 64 * 128; idx += 256) {         // W3 [128,64]
        int n = idx >> 7, k = idx & 127;
        W3T[n * M_W3T_STR + k] = (_Float16)W3[k * 64 + n];
    }
    for (int idx = tid; idx < 16 * 64; idx += 256) {          // W4 [64,10] pad->16
        int n = idx >> 6, k = idx & 63;
        W4T[n * M_W4T_STR + k] = (n < 10) ? (_Float16)W4[k * 10 + n] : (_Float16)0.f;
    }
    if (tid < 128) b2l[tid] = b2[tid];
    if (tid < 64)  b3l[tid] = b3[tid];
    if (tid < 16)  b4l[tid] = (tid < 10) ? b4[tid] : 0.f;
    __syncthreads();

    const int lane = tid & 31;
    const int wave = tid >> 5;
    const int half = lane >> 4;
    const int lm   = lane & 15;
    const int nwaves = gridDim.x * 8;

    _Float16* h2s = (_Float16*)(smem + M_SCR_OFF) + wave * M_SCR_HALVES;  // [16][136]
    _Float16* h3s = h2s + 16 * M_H2_STR;                                  // [16][72]

    const f32x8 vzero = {0.f,0.f,0.f,0.f,0.f,0.f,0.f,0.f};

    for (int t = blockIdx.x * 8 + wave; t < NTILES; t += nwaves) {
        const int q = t / 10;
        const int c = t - q * 10;
        const _Float16* qrow = QA + (size_t)q * 256;
        const _Float16* srow = SB + (size_t)(c * 16 + lm) * 256;

        // ---------------- layer 2: [16,256] x [256,128] ----------------
        f32x8 acc2[8];
#pragma unroll
        for (int j = 0; j < 8; ++j) acc2[j] = vzero;
#pragma unroll
        for (int kt = 0; kt < 8; ++kt) {
            const int k0 = kt * 32 + half * 8;
            f16x8 r0 = *(const f16x8*)(qrow + k0)      + *(const f16x8*)(srow + k0);
            f16x8 r1 = *(const f16x8*)(qrow + k0 + 16) + *(const f16x8*)(srow + k0 + 16);
#pragma unroll
            for (int i = 0; i < 8; ++i) {               // relu (v_pk_max_f16)
                r0[i] = r0[i] > (_Float16)0 ? r0[i] : (_Float16)0;
                r1[i] = r1[i] > (_Float16)0 ? r1[i] : (_Float16)0;
            }
            f16x16 a = cat16(r0, r1);
            // Preload all 8 B operands into independent registers first.
            f16x16 bmat[8];
#pragma unroll
            for (int nt = 0; nt < 8; ++nt)
                bmat[nt] = ldB(W2T + (nt * 16 + lm) * M_W2T_STR + kt * 32 + half * 8);
#pragma unroll
            for (int nt = 0; nt < 8; ++nt)
                acc2[nt] = wmma_f16(a, bmat[nt], acc2[nt]);
        }
        // bias + relu + f16, reshape D-layout -> row-major LDS
#pragma unroll
        for (int nt = 0; nt < 8; ++nt) {
            const int n = nt * 16 + lm;
            const float bb = b2l[n];
#pragma unroll
            for (int v = 0; v < 8; ++v) {
                const int m = v + half * 8;
                h2s[m * M_H2_STR + n] = (_Float16)fmaxf(acc2[nt][v] + bb, 0.f);
            }
        }

        // ---------------- layer 3: [16,128] x [128,64] ----------------
        f32x8 acc3[4];
#pragma unroll
        for (int j = 0; j < 4; ++j) acc3[j] = vzero;
        const _Float16* a3row = h2s + lm * M_H2_STR;
#pragma unroll
        for (int kt = 0; kt < 4; ++kt) {
            const int k0 = kt * 32 + half * 8;
            f16x16 a = cat16(*(const f16x8*)(a3row + k0), *(const f16x8*)(a3row + k0 + 16));
            f16x16 bmat[4];
#pragma unroll
            for (int nt = 0; nt < 4; ++nt)
                bmat[nt] = ldB(W3T + (nt * 16 + lm) * M_W3T_STR + kt * 32 + half * 8);
#pragma unroll
            for (int nt = 0; nt < 4; ++nt)
                acc3[nt] = wmma_f16(a, bmat[nt], acc3[nt]);
        }
#pragma unroll
        for (int nt = 0; nt < 4; ++nt) {
            const int n = nt * 16 + lm;
            const float bb = b3l[n];
#pragma unroll
            for (int v = 0; v < 8; ++v) {
                const int m = v + half * 8;
                h3s[m * M_H3_STR + n] = (_Float16)fmaxf(acc3[nt][v] + bb, 0.f);
            }
        }

        // ---------------- layer 4: [16,64] x [64,16(pad of 10)] ----------------
        f32x8 acc4 = vzero;
        const _Float16* a4row = h3s + lm * M_H3_STR;
        {
            f16x16 a0 = cat16(*(const f16x8*)(a4row + half * 8),
                              *(const f16x8*)(a4row + half * 8 + 16));
            f16x16 a1 = cat16(*(const f16x8*)(a4row + 32 + half * 8),
                              *(const f16x8*)(a4row + 32 + half * 8 + 16));
            f16x16 b0 = ldB(W4T + lm * M_W4T_STR + half * 8);
            f16x16 b1 = ldB(W4T + lm * M_W4T_STR + 32 + half * 8);
            acc4 = wmma_f16(a0, b0, acc4);
            acc4 = wmma_f16(a1, b1, acc4);
        }

        // tanh, mean over 10 outputs, sum over 16 support rows.
        float partial = 0.f;
        if (lm < 10) {
            const float bb = b4l[lm];
#pragma unroll
            for (int v = 0; v < 8; ++v) partial += tanhf(acc4[v] + bb);
        }
#pragma unroll
        for (int off = 16; off > 0; off >>= 1)
            partial += __shfl_xor(partial, off, 32);
        if (lane == 0) out[t] = partial * 0.1f;
    }
}

// ---------------------------------------------------------------------------
extern "C" void kernel_launch(void* const* d_in, const int* in_sizes, int n_in,
                              void* d_out, int out_size, void* d_ws, size_t ws_size,
                              hipStream_t stream) {
    (void)in_sizes; (void)n_in; (void)out_size; (void)ws_size;
    const float* queries  = (const float*)d_in[0];   // [2048,256]
    const float* supports = (const float*)d_in[1];   // [10,16,256] -> [160,256]
    const float* W1 = (const float*)d_in[2];         // [512,256]
    const float* b1 = (const float*)d_in[3];
    const float* W2 = (const float*)d_in[4];         // [256,128]
    const float* b2 = (const float*)d_in[5];
    const float* W3 = (const float*)d_in[6];         // [128,64]
    const float* b3 = (const float*)d_in[7];
    const float* W4 = (const float*)d_in[8];         // [64,10]
    const float* b4 = (const float*)d_in[9];
    float* out = (float*)d_out;                      // [2048,10]

    _Float16* QA = (_Float16*)d_ws;                  // 2048*256 f16 = 1 MB
    _Float16* SB = QA + 2048 * 256;                  //  160*256 f16

    hipFuncSetAttribute((const void*)gemm256_f16out,
                        hipFuncAttributeMaxDynamicSharedMemorySize, P_SMEM_SIZE);
    hipFuncSetAttribute((const void*)simnet_main,
                        hipFuncAttributeMaxDynamicSharedMemorySize, M_SMEM_SIZE);

    // QA = queries @ W1[:256,:]         (bias folded into SB instead)
    gemm256_f16out<<<16, 256, P_SMEM_SIZE, stream>>>(queries, W1, nullptr, QA, 2048);
    // SB = supports @ W1[256:,:] + b1
    gemm256_f16out<<<2, 256, P_SMEM_SIZE, stream>>>(supports, W1 + 256 * 256, b1, SB, 160);
    // Main fused MLP + reduction.
    simnet_main<<<320, 256, M_SMEM_SIZE, stream>>>(QA, SB, W2, b2, W3, b3, W4, b4, out);
}
